// BasicGNN_91182155694567
// MI455X (gfx1250) — compile-verified, matrix operand order checked by reading, not run on hardware
//
#include <hip/hip_runtime.h>
#include <hip/hip_bf16.h>

typedef __attribute__((ext_vector_type(16))) __bf16 v16bf;
typedef __attribute__((ext_vector_type(8)))  float  v8f;

#define IN_C  64
#define H_C   128
#define OUT_C 10
#define G_C   64

// ---------------------------------------------------------------------------
// init: t0 = 2*x (fold self-loop + residual), zero pooled accumulators
// ---------------------------------------------------------------------------
__global__ void k_init0(const float* __restrict__ x, float* __restrict__ t0,
                        float* __restrict__ pooled, float* __restrict__ counts,
                        long n_elems) {
    long gid = (long)blockIdx.x * blockDim.x + threadIdx.x;
    if (gid < (long)G_C * H_C) pooled[gid] = 0.0f;
    if (gid < G_C)             counts[gid] = 0.0f;
    if (gid < n_elems)         t0[gid] = 2.0f * x[gid];
}

// t = 2*h  (second message-passing init)
__global__ void k_scale2(const float* __restrict__ h, float* __restrict__ t,
                         long n_elems) {
    long gid = (long)blockIdx.x * blockDim.x + threadIdx.x;
    if (gid < n_elems) t[gid] = 2.0f * h[gid];
}

// ---------------------------------------------------------------------------
// Edge scatter: dst[row[e]] += src[col[e]]  (vector of F floats per edge)
// Consecutive lanes = consecutive features of one edge -> coalesced.
// ---------------------------------------------------------------------------
template<int F>
__global__ void k_scatter(const float* __restrict__ src, float* __restrict__ dst,
                          const int* __restrict__ row, const int* __restrict__ col,
                          int E) {
    long gid = (long)blockIdx.x * blockDim.x + threadIdx.x;
    long total = (long)E * F;
    if (gid >= total) return;
    int e = (int)(gid / F);
    int f = (int)(gid % F);
    atomicAdd(&dst[(long)row[e] * F + f], src[(long)col[e] * F + f]);
}

// ---------------------------------------------------------------------------
// WMMA GEMM + bias + ReLU:  Out[N,128] = relu(A[N,K] @ W[K,128] + bias)
// One block = 16 rows; wave w handles columns [16w, 16w+16).
// Fragment layouts per CDNA5 ISA 7.12.2 (16-bit A 16x32, f32 C/D 16x16).
// ---------------------------------------------------------------------------
template<int K>
__global__ __launch_bounds__(256) void k_gemm_bias_relu(
    const float* __restrict__ A, const float* __restrict__ W,
    const float* __restrict__ bias, float* __restrict__ Out) {
    const int lane = threadIdx.x & 31;
    const int wave = threadIdx.x >> 5;
    const int half = lane >> 4;     // 0: lanes 0-15, 1: lanes 16-31
    const int mr   = lane & 15;     // A row / B column within tile
    const int m0   = blockIdx.x * 16;
    const int n0   = wave * 16;

    v8f acc = {};
#pragma unroll
    for (int ks = 0; ks < K / 32; ++ks) {
        v16bf a, b;
#pragma unroll
        for (int v = 0; v < 8; ++v) {
            // K index of packed pair {2v, 2v+1} for this half-wave (ISA layout)
            int kb = (v < 4) ? (half * 8 + 2 * v) : (16 + half * 8 + 2 * (v - 4));
            int k  = ks * 32 + kb;
            a[2 * v]     = (__bf16)A[(long)(m0 + mr) * K + k];
            a[2 * v + 1] = (__bf16)A[(long)(m0 + mr) * K + k + 1];
            b[2 * v]     = (__bf16)W[(long)k * H_C + n0 + mr];
            b[2 * v + 1] = (__bf16)W[(long)(k + 1) * H_C + n0 + mr];
        }
        acc = __builtin_amdgcn_wmma_f32_16x16x32_bf16(
            /*neg_a=*/false, a, /*neg_b=*/false, b,
            /*c_mod=*/(short)0, acc, /*reuse_a=*/false, /*reuse_b=*/false);
    }
    // D layout: lanes 0-15 -> M=v,   N=lane;  lanes 16-31 -> M=v+8, N=lane-16
    const int ncol = lane & 15;
    const float bv = bias[n0 + ncol];
#pragma unroll
    for (int v = 0; v < 8; ++v) {
        int m = m0 + v + 8 * half;
        float val = acc[v] + bv;
        Out[(long)m * H_C + n0 + ncol] = fmaxf(val, 0.0f);
    }
}

// Scalar tail for N % 16 != 0 (not hit for N=50000, kept for generality)
__global__ void k_gemm_tail(const float* __restrict__ A, const float* __restrict__ W,
                            const float* __restrict__ bias, float* __restrict__ Out,
                            int K, int mstart, int mcount) {
    int gid = blockIdx.x * blockDim.x + threadIdx.x;
    if (gid >= mcount * H_C) return;
    int m = mstart + gid / H_C;
    int n = gid % H_C;
    float s = bias[n];
    for (int k = 0; k < K; ++k) s += A[(long)m * K + k] * W[(long)k * H_C + n];
    Out[(long)m * H_C + n] = fmaxf(s, 0.0f);
}

// ---------------------------------------------------------------------------
// Pooling
// ---------------------------------------------------------------------------
__global__ void k_pool(const float* __restrict__ h2, const int* __restrict__ batch,
                       float* __restrict__ pooled, int N) {
    long gid = (long)blockIdx.x * blockDim.x + threadIdx.x;
    long total = (long)N * H_C;
    if (gid >= total) return;
    int i = (int)(gid / H_C);
    int f = (int)(gid % H_C);
    atomicAdd(&pooled[(long)batch[i] * H_C + f], h2[gid]);
}

__global__ void k_counts(const int* __restrict__ batch, float* __restrict__ counts,
                         int N) {
    int i = blockIdx.x * blockDim.x + threadIdx.x;
    if (i < N) atomicAdd(&counts[batch[i]], 1.0f);
}

// out[G,10] = (pooled / counts) @ Wout + bout
__global__ void k_out(const float* __restrict__ pooled, const float* __restrict__ counts,
                      const float* __restrict__ Wout, const float* __restrict__ bout,
                      float* __restrict__ out) {
    int gid = blockIdx.x * blockDim.x + threadIdx.x;
    if (gid >= G_C * OUT_C) return;
    int g = gid / OUT_C;
    int o = gid % OUT_C;
    float inv = 1.0f / fmaxf(counts[g], 1.0f);
    float s = bout[o];
    for (int k = 0; k < H_C; ++k)
        s += pooled[(long)g * H_C + k] * inv * Wout[(long)k * OUT_C + o];
    out[gid] = s;
}

// ---------------------------------------------------------------------------
extern "C" void kernel_launch(void* const* d_in, const int* in_sizes, int n_in,
                              void* d_out, int out_size, void* d_ws, size_t ws_size,
                              hipStream_t stream) {
    const float* x     = (const float*)d_in[0];
    const int*   ei    = (const int*)  d_in[1];
    const int*   batch = (const int*)  d_in[2];
    const float* W1    = (const float*)d_in[3];
    const float* b1    = (const float*)d_in[4];
    const float* W2    = (const float*)d_in[5];
    const float* b2    = (const float*)d_in[6];
    const float* Wout  = (const float*)d_in[7];
    const float* bout  = (const float*)d_in[8];
    float* out = (float*)d_out;

    const int N = in_sizes[0] / IN_C;
    const int E = in_sizes[1] / 2;
    const int* row = ei;
    const int* col = ei + E;

    // workspace layout (floats)
    float* t0     = (float*)d_ws;                 // [N, 64]
    float* hbuf   = t0   + (size_t)N * IN_C;      // [N, 128]  h1, then reused as h2
    float* t1     = hbuf + (size_t)N * H_C;       // [N, 128]
    float* pooled = t1   + (size_t)N * H_C;       // [G, 128]
    float* counts = pooled + (size_t)G_C * H_C;   // [G]

    const int B = 256;
    auto blocks = [](long n, int b) { return (unsigned)((n + b - 1) / b); };

    // ---- layer 1: mp(x) -> t0 ----
    long n0e = (long)N * IN_C;
    k_init0<<<blocks(n0e, B), B, 0, stream>>>(x, t0, pooled, counts, n0e);
    k_scatter<IN_C><<<blocks((long)E * IN_C, B), B, 0, stream>>>(x, t0, row, col, E);

    // ---- h1 = relu(t0 @ W1 + b1) ----
    int mtiles = N / 16, mrem = N % 16;
    k_gemm_bias_relu<IN_C><<<mtiles, B, 0, stream>>>(t0, W1, b1, hbuf);
    if (mrem)
        k_gemm_tail<<<blocks((long)mrem * H_C, B), B, 0, stream>>>(
            t0, W1, b1, hbuf, IN_C, mtiles * 16, mrem);

    // ---- layer 2: mp(h1) -> t1 ----
    long n1e = (long)N * H_C;
    k_scale2<<<blocks(n1e, B), B, 0, stream>>>(hbuf, t1, n1e);
    k_scatter<H_C><<<blocks((long)E * H_C, B), B, 0, stream>>>(hbuf, t1, row, col, E);

    // ---- h2 = relu(t1 @ W2 + b2) (reuse hbuf) ----
    k_gemm_bias_relu<H_C><<<mtiles, B, 0, stream>>>(t1, W2, b2, hbuf);
    if (mrem)
        k_gemm_tail<<<blocks((long)mrem * H_C, B), B, 0, stream>>>(
            t1, W2, b2, hbuf, H_C, mtiles * 16, mrem);

    // ---- pooling + head ----
    k_pool<<<blocks(n1e, B), B, 0, stream>>>(hbuf, batch, pooled, N);
    k_counts<<<blocks(N, B), B, 0, stream>>>(batch, counts, N);
    k_out<<<blocks(G_C * OUT_C, B), B, 0, stream>>>(pooled, counts, Wout, bout, out);
}
